// AdsorptionGNN_43817256354335
// MI455X (gfx1250) — matrix operand order repaired and checked by reading, not compiled, not run on hardware
//
#include <hip/hip_runtime.h>

// ---------------------------------------------------------------------------
// AdsorptionGNN on MI455X (gfx1250): fused edge-MLP + scatter via WMMA f16.
// A-operands are wave-private -> fragments gathered directly from global
// (L2-resident). Edge kernel uses 2x M-register-blocking: each wave owns 32
// edge rows so every B fragment feeds two WMMAs (halves B cache traffic).
// ---------------------------------------------------------------------------

#define N_NODES    50000
#define N_EDGES    800000
#define HID        128
#define EDGE_DIM   4
#define LAYERS     4
#define NUM_GRAPHS 512
#define K1         260     // 2*HID + EDGE_DIM
#define K1P        288     // padded to multiple of 32
#define K2         128
#define KN         256
#define HPAD       (HID + 8)   // half_t row stride: 272B -> rotates LDS banks
#define ETILE      256          // edges per block (8 waves x 32 rows)

typedef _Float16 half_t;
typedef _Float16 v16h __attribute__((ext_vector_type(16)));
typedef _Float16 v8h  __attribute__((ext_vector_type(8)));
typedef float    v8f  __attribute__((ext_vector_type(8)));

__device__ __forceinline__ float silu_f(float x) { return x / (1.0f + __expf(-x)); }

__device__ __forceinline__ v8f wmma_f16(v16h a, v16h b, v8f c) {
  // D = A(16x32,f16) * B(32x16,f16) + C(16x16,f32)
  return __builtin_amdgcn_wmma_f32_16x16x32_f16(false, a, false, b, (short)0, c,
                                                false, false);
}

// A fragment from a per-lane row pointer (rowp already includes the lane's
// kb8 = (lane>>4)*8 half offset). ISA 16-bit A layout:
//  lanes 0-15 (M=lane):  elems 0-7 = K k0..k0+7,  elems 8-15 = K k0+16..k0+23
//  lanes 16-31:          same with +8 K offset (folded into rowp).
__device__ __forceinline__ v16h frag_a_h(const half_t* rowp, int koff) {
  v8h lo = *(const v8h*)(rowp + koff);
  v8h hi = *(const v8h*)(rowp + koff + 16);
  return __builtin_shufflevector(lo, hi, 0,1,2,3,4,5,6,7,8,9,10,11,12,13,14,15);
}

// Same, but source row is f32 (converted in registers), optionally scaled.
__device__ __forceinline__ v16h frag_a_f32(const float* rowp, int koff, float scale) {
  v16h a;
  #pragma unroll
  for (int q = 0; q < 8; ++q) a[q]     = (half_t)(rowp[koff + q] * scale);
  #pragma unroll
  for (int q = 0; q < 8; ++q) a[8 + q] = (half_t)(rowp[koff + 16 + q] * scale);
  return a;
}

// B fragment: 32x16 slice of W^T stored [nOut][kIn] row-major (stride kp).
// B layout (matches A): lanes 0-15 (N=lane): elems 0-7 = K k0..k0+7, elems 8-15 = K k0+16..k0+23; lanes 16-31: +8.
__device__ __forceinline__ v16h frag_b(const half_t* __restrict__ wt, int kp,
                                       int ncol, int k0, int lane) {
  int n  = lane & 15;
  int kb = (lane >> 4) << 3;
  const half_t* p = wt + (size_t)(ncol + n) * kp + k0 + kb;
  v8h lo = *(const v8h*)(p);
  v8h hi = *(const v8h*)(p + 16);
  return __builtin_shufflevector(lo, hi, 0,1,2,3,4,5,6,7,8,9,10,11,12,13,14,15);
}

// ---------------------------------------------------------------------------
__global__ void zero_kernel(float* __restrict__ p, int n) {
  int i = blockIdx.x * blockDim.x + threadIdx.x;
  if (i < n) p[i] = 0.0f;
}

// W [K,N] f32 -> Wt [N,Kp] f16, zero-padded in K.
__global__ void wprep_kernel(const float* __restrict__ W, half_t* __restrict__ Wt,
                             int K, int N, int Kp) {
  int idx = blockIdx.x * blockDim.x + threadIdx.x;
  if (idx >= N * Kp) return;
  int n = idx / Kp, k = idx % Kp;
  Wt[idx] = (k < K) ? (half_t)W[(size_t)k * N + n] : (half_t)0.f;
}

__global__ void embed_kernel(const int* __restrict__ z, const float* __restrict__ emb,
                             half_t* __restrict__ x) {
  int idx = blockIdx.x * blockDim.x + threadIdx.x;
  if (idx >= N_NODES * HID) return;
  int n = idx >> 7, c = idx & 127;
  x[idx] = (half_t)emb[(size_t)z[n] * HID + c];
}

// ---------------------------------------------------------------------------
// Edge kernel: 256 edges/block, 8 waves, 32 edge-rows (2 M-tiles) per wave.
// h = silu([x_src|x_dst|ea] @ W1 + b1); msg = silu(h @ W2 + b2); agg[dst] += msg
__global__ __launch_bounds__(256)
void edge_kernel(const half_t* __restrict__ x,
                 const int* __restrict__ src, const int* __restrict__ dst,
                 const float* __restrict__ ea,
                 const half_t* __restrict__ w1t, const float* __restrict__ b1,
                 const half_t* __restrict__ w2t, const float* __restrict__ b2,
                 float* __restrict__ agg) {
  __shared__ half_t hT[ETILE][HPAD];   // ~68 KB: C->A layout transpose of h

  const int tid  = threadIdx.x;
  const int lane = tid & 31;
  const int wave = tid >> 5;
  const int e0   = blockIdx.x * ETILE;
  const int m0   = wave * 32;           // 2 M-tiles: rows m0..m0+31
  const int mrow = lane & 15;
  const int kb8  = (lane >> 4) << 3;
  const int er0  = e0 + m0 + mrow;      // A row for M-tile 0
  const int er1  = er0 + 16;            // A row for M-tile 1

  // Per-lane A row pointers (random gather, L2-resident x).
  const half_t* sp0 = x + (size_t)src[er0] * HID + kb8;
  const half_t* sp1 = x + (size_t)src[er1] * HID + kb8;
  const half_t* dp0 = x + (size_t)dst[er0] * HID + kb8;
  const half_t* dp1 = x + (size_t)dst[er1] * HID + kb8;

  // Attr fragments (K tile 256..287): only lanes 0-15, elems 0-3 carry data.
  v16h aAt0, aAt1;
  #pragma unroll
  for (int q = 0; q < 16; ++q) { aAt0[q] = (half_t)0.f; aAt1[q] = (half_t)0.f; }
  if (lane < 16) {
    const float* a0 = ea + (size_t)er0 * EDGE_DIM;
    const float* a1 = ea + (size_t)er1 * EDGE_DIM;
    #pragma unroll
    for (int q = 0; q < EDGE_DIM; ++q) {
      aAt0[q] = (half_t)a0[q];
      aAt1[q] = (half_t)a1[q];
    }
  }

  // ---- GEMM1: [32 x 288] @ [288 x 128], B reused across both M-tiles ----
  v8f acc0[8] = {}, acc1[8] = {};
  for (int kt = 0; kt < 4; ++kt) {               // x[src] K region 0..127
    v16h a0 = frag_a_h(sp0, kt * 32);
    v16h a1 = frag_a_h(sp1, kt * 32);
    #pragma unroll
    for (int nt = 0; nt < 8; ++nt) {
      v16h b = frag_b(w1t, K1P, nt * 16, kt * 32, lane);
      acc0[nt] = wmma_f16(a0, b, acc0[nt]);
      acc1[nt] = wmma_f16(a1, b, acc1[nt]);
    }
  }
  for (int kt = 0; kt < 4; ++kt) {               // x[dst] K region 128..255
    v16h a0 = frag_a_h(dp0, kt * 32);
    v16h a1 = frag_a_h(dp1, kt * 32);
    #pragma unroll
    for (int nt = 0; nt < 8; ++nt) {
      v16h b = frag_b(w1t, K1P, nt * 16, 128 + kt * 32, lane);
      acc0[nt] = wmma_f16(a0, b, acc0[nt]);
      acc1[nt] = wmma_f16(a1, b, acc1[nt]);
    }
  }
  #pragma unroll
  for (int nt = 0; nt < 8; ++nt) {               // edge_attr K region 256..287
    v16h b = frag_b(w1t, K1P, nt * 16, 256, lane);
    acc0[nt] = wmma_f16(aAt0, b, acc0[nt]);
    acc1[nt] = wmma_f16(aAt1, b, acc1[nt]);
  }

  // bias + silu, transpose C layout -> row-major h in LDS
  const int c  = lane & 15;
  const int mh = (lane >> 4) << 3;
  #pragma unroll
  for (int nt = 0; nt < 8; ++nt) {
    float bias = b1[nt * 16 + c];
    #pragma unroll
    for (int r = 0; r < 8; ++r) {
      hT[m0 + mh + r][nt * 16 + c]      = (half_t)silu_f(acc0[nt][r] + bias);
      hT[m0 + 16 + mh + r][nt * 16 + c] = (half_t)silu_f(acc1[nt][r] + bias);
    }
  }
  __syncthreads();   // ordering fence for cross-lane LDS reads

  // ---- GEMM2: [32 x 128] @ [128 x 128] ----
  const half_t* hr0 = &hT[m0 + mrow][kb8];
  const half_t* hr1 = &hT[m0 + 16 + mrow][kb8];
  v8f m0acc[8] = {}, m1acc[8] = {};
  for (int kt = 0; kt < K2 / 32; ++kt) {
    v16h a0 = frag_a_h(hr0, kt * 32);
    v16h a1 = frag_a_h(hr1, kt * 32);
    #pragma unroll
    for (int nt = 0; nt < 8; ++nt) {
      v16h b = frag_b(w2t, K2, nt * 16, kt * 32, lane);
      m0acc[nt] = wmma_f16(a0, b, m0acc[nt]);
      m1acc[nt] = wmma_f16(a1, b, m1acc[nt]);
    }
  }

  // Fused scatter-add of messages into agg[dst] (f32 L2 atomics).
  int dr0[8], dr1[8];
  #pragma unroll
  for (int r = 0; r < 8; ++r) {
    dr0[r] = dst[e0 + m0 + mh + r];
    dr1[r] = dst[e0 + m0 + 16 + mh + r];
  }
  #pragma unroll
  for (int nt = 0; nt < 8; ++nt) {
    float bias = b2[nt * 16 + c];
    #pragma unroll
    for (int r = 0; r < 8; ++r) {
      float v0 = silu_f(m0acc[nt][r] + bias);
      float v1 = silu_f(m1acc[nt][r] + bias);
      atomicAdd(&agg[(size_t)dr0[r] * HID + nt * 16 + c], v0);
      atomicAdd(&agg[(size_t)dr1[r] * HID + nt * 16 + c], v1);
    }
  }
}

// ---------------------------------------------------------------------------
// Node kernel: x' = silu([x | agg] @ Wn + bn). 128 nodes/block, zero LDS.
__global__ __launch_bounds__(256)
void node_kernel(const half_t* __restrict__ xin, const float* __restrict__ agg,
                 const half_t* __restrict__ wt, const float* __restrict__ bias,
                 half_t* __restrict__ xout, int nNodes) {
  const int tid = threadIdx.x, lane = tid & 31, wave = tid >> 5;
  const int n0 = blockIdx.x * 128, m0 = wave * 16;
  const int mrow = lane & 15;
  const int kb8  = (lane >> 4) << 3;
  int row  = n0 + m0 + mrow;
  int rowc = row < nNodes ? row : nNodes - 1;    // clamp: safe loads, guarded store

  const half_t* xr = xin + (size_t)rowc * HID + kb8;
  const float*  ar = agg + (size_t)rowc * HID + kb8;

  v8f acc[8] = {};
  for (int kt = 0; kt < 4; ++kt) {               // x K region 0..127
    v16h a = frag_a_h(xr, kt * 32);
    #pragma unroll
    for (int nt = 0; nt < 8; ++nt)
      acc[nt] = wmma_f16(a, frag_b(wt, KN, nt * 16, kt * 32, lane), acc[nt]);
  }
  for (int kt = 0; kt < 4; ++kt) {               // agg K region 128..255
    v16h a = frag_a_f32(ar, kt * 32, 1.0f);
    #pragma unroll
    for (int nt = 0; nt < 8; ++nt)
      acc[nt] = wmma_f16(a, frag_b(wt, KN, nt * 16, 128 + kt * 32, lane), acc[nt]);
  }

  const int c = lane & 15, mh = (lane >> 4) << 3;
  #pragma unroll
  for (int nt = 0; nt < 8; ++nt) {
    float bv = bias[nt * 16 + c];
    #pragma unroll
    for (int r = 0; r < 8; ++r) {
      int m = n0 + m0 + mh + r;
      if (m < nNodes)
        xout[(size_t)m * HID + nt * 16 + c] = (half_t)silu_f(acc[nt][r] + bv);
    }
  }
}

// ---------------------------------------------------------------------------
__global__ void pool_kernel(const half_t* __restrict__ x, const int* __restrict__ batch,
                            float* __restrict__ sums, float* __restrict__ counts) {
  int idx = blockIdx.x * blockDim.x + threadIdx.x;
  if (idx >= N_NODES * HID) return;
  int n = idx >> 7, c = idx & 127;
  int g = batch[n];
  atomicAdd(&sums[(size_t)g * HID + c], (float)x[idx]);
  if (c == 0) atomicAdd(&counts[g], 1.0f);
}

// Readout: g = sums/counts; pred = silu(g@rw1+rb1)@rw2 + rb2. 128 graphs/block.
__global__ __launch_bounds__(256)
void readout_kernel(const float* __restrict__ sums, const float* __restrict__ counts,
                    const half_t* __restrict__ rw1t, const float* __restrict__ rb1,
                    const float* __restrict__ rw2, const float* __restrict__ rb2,
                    float* __restrict__ out) {
  __shared__ float sT[128][HID + 1];   // +1 float: rotate banks
  const int tid = threadIdx.x, lane = tid & 31, wave = tid >> 5;
  const int g0 = blockIdx.x * 128, m0 = wave * 16;
  const int mrow = lane & 15;
  const int kb8  = (lane >> 4) << 3;
  const int grow = g0 + m0 + mrow;

  float cnt = counts[grow];
  float inv = 1.0f / (cnt < 1.0f ? 1.0f : cnt);
  const float* gr = sums + (size_t)grow * HID + kb8;

  v8f acc[8] = {};
  for (int kt = 0; kt < HID / 32; ++kt) {
    v16h a = frag_a_f32(gr, kt * 32, inv);       // mean fused into A build
    #pragma unroll
    for (int nt = 0; nt < 8; ++nt)
      acc[nt] = wmma_f16(a, frag_b(rw1t, HID, nt * 16, kt * 32, lane), acc[nt]);
  }
  const int c = lane & 15, mh = (lane >> 4) << 3;
  #pragma unroll
  for (int nt = 0; nt < 8; ++nt) {
    float bv = rb1[nt * 16 + c];
    #pragma unroll
    for (int r = 0; r < 8; ++r)
      sT[m0 + mh + r][nt * 16 + c] = silu_f(acc[nt][r] + bv);
  }
  __syncthreads();

  if (tid < 128) {
    float s = rb2[0];
    for (int cc = 0; cc < HID; ++cc) s += sT[tid][cc] * rw2[cc];
    out[g0 + tid] = s;
  }
}

// ---------------------------------------------------------------------------
extern "C" void kernel_launch(void* const* d_in, const int* in_sizes, int n_in,
                              void* d_out, int out_size, void* d_ws, size_t ws_size,
                              hipStream_t stream) {
  (void)in_sizes; (void)n_in; (void)out_size; (void)ws_size;
  const int*   z     = (const int*)  d_in[0];
  const int*   ei    = (const int*)  d_in[1];
  const float* ea    = (const float*)d_in[2];
  const int*   batch = (const int*)  d_in[3];
  const float* emb   = (const float*)d_in[4];
  const float* ew1   = (const float*)d_in[5];
  const float* eb1   = (const float*)d_in[6];
  const float* ew2   = (const float*)d_in[7];
  const float* eb2   = (const float*)d_in[8];
  const float* nw    = (const float*)d_in[9];
  const float* nb    = (const float*)d_in[10];
  const float* rw1   = (const float*)d_in[11];
  const float* rb1   = (const float*)d_in[12];
  const float* rw2   = (const float*)d_in[13];
  const float* rb2   = (const float*)d_in[14];
  float* out = (float*)d_out;
  const int* src = ei;
  const int* dst = ei + N_EDGES;

  char* ws = (char*)d_ws;
  size_t off = 0;
  auto take = [&](size_t bytes) -> char* {
    char* p = ws + off;
    off += (bytes + 255) & ~(size_t)255;
    return p;
  };
  half_t* xA   = (half_t*)take((size_t)N_NODES * HID * 2);
  half_t* xB   = (half_t*)take((size_t)N_NODES * HID * 2);
  float*  agg  = (float*) take((size_t)N_NODES * HID * 4);
  half_t* ew1t = (half_t*)take((size_t)LAYERS * HID * K1P * 2);
  half_t* ew2t = (half_t*)take((size_t)LAYERS * HID * K2 * 2);
  half_t* nwt  = (half_t*)take((size_t)LAYERS * HID * KN * 2);
  half_t* rw1t = (half_t*)take((size_t)HID * HID * 2);
  float*  sums = (float*) take((size_t)NUM_GRAPHS * HID * 4);
  float*  cnts = (float*) take((size_t)NUM_GRAPHS * 4);

  // Weight prep: f32 [K,N] -> f16 [N,Kpad] transposed, zero-padded.
  for (int l = 0; l < LAYERS; ++l) {
    wprep_kernel<<<(HID * K1P + 255) / 256, 256, 0, stream>>>(
        ew1 + (size_t)l * K1 * HID, ew1t + (size_t)l * HID * K1P, K1, HID, K1P);
    wprep_kernel<<<(HID * K2 + 255) / 256, 256, 0, stream>>>(
        ew2 + (size_t)l * K2 * HID, ew2t + (size_t)l * HID * K2, K2, HID, K2);
    wprep_kernel<<<(HID * KN + 255) / 256, 256, 0, stream>>>(
        nw + (size_t)l * KN * HID, nwt + (size_t)l * HID * KN, KN, HID, KN);
  }
  wprep_kernel<<<(HID * HID + 255) / 256, 256, 0, stream>>>(rw1, rw1t, HID, HID, HID);

  embed_kernel<<<(N_NODES * HID + 255) / 256, 256, 0, stream>>>(z, emb, xA);

  half_t* xc = xA;
  half_t* xn = xB;
  for (int l = 0; l < LAYERS; ++l) {
    zero_kernel<<<(N_NODES * HID + 255) / 256, 256, 0, stream>>>(agg, N_NODES * HID);
    edge_kernel<<<N_EDGES / ETILE, 256, 0, stream>>>(
        xc, src, dst, ea,
        ew1t + (size_t)l * HID * K1P, eb1 + (size_t)l * HID,
        ew2t + (size_t)l * HID * K2,  eb2 + (size_t)l * HID, agg);
    node_kernel<<<(N_NODES + 127) / 128, 256, 0, stream>>>(
        xc, agg, nwt + (size_t)l * HID * KN, nb + (size_t)l * HID, xn, N_NODES);
    half_t* t = xc; xc = xn; xn = t;
  }

  zero_kernel<<<(NUM_GRAPHS * HID + 255) / 256, 256, 0, stream>>>(sums, NUM_GRAPHS * HID);
  zero_kernel<<<(NUM_GRAPHS + 255) / 256, 256, 0, stream>>>(cnts, NUM_GRAPHS);
  pool_kernel<<<(N_NODES * HID + 255) / 256, 256, 0, stream>>>(xc, batch, sums, cnts);
  readout_kernel<<<NUM_GRAPHS / 128, 256, 0, stream>>>(sums, cnts, rw1t, rb1, rw2, rb2, out);
}
